// GAT_REDDIT_51118700757723
// MI455X (gfx1250) — matrix operand order repaired and checked by reading, not compiled, not run on hardware
//
#include <hip/hip_runtime.h>
#include <hip/hip_bf16.h>

// ---------------------------------------------------------------------------
// CDNA5 (gfx1250) 2-layer GAT.  Heavy GEMMs run through v_wmma_f32_16x16x32_bf16
// (fp32 inputs converted to bf16 tiles in LDS, fp32 accumulate), double-buffered
// LDS staging with branch-free boundary handling.  Edge softmax / aggregation
// are memory-bound fp32 atomics.
// ---------------------------------------------------------------------------

typedef __attribute__((ext_vector_type(16))) __bf16 v16bf;
typedef __attribute__((ext_vector_type(8)))  float  v8f;

union FragBF { v16bf v; unsigned u[8]; };
union FragF  { v8f   v; float    f[8]; };

__device__ __forceinline__ unsigned short f2bf(float f) {
  // round-to-nearest-even fp32 -> bf16
  unsigned u = __float_as_uint(f);
  unsigned r = u + 0x7FFFu + ((u >> 16) & 1u);
  return (unsigned short)(r >> 16);
}

__device__ __forceinline__ float waveSum(float v) {
#pragma unroll
  for (int o = 16; o > 0; o >>= 1) v += __shfl_xor(v, o, 32);
  return v;
}
__device__ __forceinline__ float waveMax(float v) {
#pragma unroll
  for (int o = 16; o > 0; o >>= 1) v = fmaxf(v, __shfl_xor(v, o, 32));
  return v;
}

// float atomic-max via ordered integer bit tricks (works with -FLT_MAX init)
__device__ __forceinline__ void atomicMaxF(float* addr, float v) {
  if (v >= 0.f) atomicMax((int*)addr, __float_as_int(v));
  else          atomicMin((unsigned int*)addr, __float_as_uint(v));
}

// ---------------------------------------------------------------------------
// bf16 WMMA GEMM: C[M,N] = A[M,K] * B[K,N]   (A,B,C fp32 in memory)
// block tile 128x128, BK=32, 8 waves (2x4), each wave 64x32 -> 8 wmma tiles.
// Double-buffered LDS; vectorized staging for interior tiles; boundary tiles
// use clamped-address loads + select (no exec divergence in the hot loop).
// ---------------------------------------------------------------------------
#define BM 128
#define BN 128
#define BK 32
#define LDS_S 36   // bf16 elements per LDS row (32 + pad, keeps dwords aligned)

__global__ __launch_bounds__(256, 4)   // cap at 256 VGPRs (no MSB switching)
void gat_gemm_bf16(const float* __restrict__ A, const float* __restrict__ B,
                   float* __restrict__ C, int M, int N, int K)
{
  __shared__ unsigned short sA[2][BM * LDS_S];       // [m][k]
  __shared__ unsigned short sB[2][BN * LDS_S];       // [n][k]  (transposed)

  const int tid  = threadIdx.x;
  const int wave = tid >> 5;
  const int lane = tid & 31;
  const int half = lane >> 4;       // K-half selector for 16-bit fragments
  const int ml   = lane & 15;
  const int wm   = wave & 1;        // 64-row slab
  const int wn   = wave >> 1;       // 32-col slab
  const int bm0  = blockIdx.y * BM;
  const int bn0  = blockIdx.x * BN;

  const bool vecA = ((K & 1) == 0) && (bm0 + BM <= M);   // float2-aligned rows
  const bool vecB = ((N & 3) == 0) && (bn0 + BN <= N);   // float4-aligned rows

  // ---- tile staging --------------------------------------------------------
  auto stageA = [&](int k0, int buf) {
    if (vecA && (k0 + BK <= K)) {
      // interior: 8 x (float2 load -> packed bf16x2 -> ds_store_b32)
#pragma unroll
      for (int i = 0; i < 8; ++i) {
        int idx2 = i * 256 + tid;              // bf16-pair index
        int r  = idx2 >> 4;                    // 0..127
        int c2 = (idx2 & 15) << 1;             // even column
        float2 f = *(const float2*)&A[(long long)(bm0 + r) * K + k0 + c2];
        unsigned p = (unsigned)f2bf(f.x) | ((unsigned)f2bf(f.y) << 16);
        *(unsigned*)&sA[buf][r * LDS_S + c2] = p;
      }
    } else {
      // boundary: clamped-address load + select-zero (branch-free)
#pragma unroll
      for (int i = 0; i < 16; ++i) {
        int idx = i * 256 + tid;
        int r = idx >> 5, c = idx & 31;
        int gr = bm0 + r, gc = k0 + c;
        bool ok = (gr < M) && (gc < K);
        int cr = gr < M ? gr : 0;
        int cc = gc < K ? gc : 0;
        float v = A[(long long)cr * K + cc];
        sA[buf][r * LDS_S + c] = f2bf(ok ? v : 0.f);
      }
    }
  };
  auto stageB = [&](int k0, int buf) {
    if (vecB && (k0 + BK <= K)) {
      // interior: 4 x (float4 load along N -> 4 transposed ds_store_b16)
#pragma unroll
      for (int i = 0; i < 4; ++i) {
        int idx4 = i * 256 + tid;
        int rk = idx4 >> 5;                    // 0..31
        int c4 = (idx4 & 31) << 2;
        float4 f = *(const float4*)&B[(long long)(k0 + rk) * N + bn0 + c4];
        sB[buf][(c4 + 0) * LDS_S + rk] = f2bf(f.x);
        sB[buf][(c4 + 1) * LDS_S + rk] = f2bf(f.y);
        sB[buf][(c4 + 2) * LDS_S + rk] = f2bf(f.z);
        sB[buf][(c4 + 3) * LDS_S + rk] = f2bf(f.w);
      }
    } else {
#pragma unroll
      for (int i = 0; i < 16; ++i) {
        int idx = i * 256 + tid;
        int r = idx >> 7, c = idx & 127;
        int gr = k0 + r, gc = bn0 + c;
        bool ok = (gr < K) && (gc < N);
        int cr = gr < K ? gr : 0;
        int cc = gc < N ? gc : 0;
        float v = B[(long long)cr * N + cc];
        sB[buf][c * LDS_S + r] = f2bf(ok ? v : 0.f);
      }
    }
  };

  v8f acc[4][2] = {};

  const int nk = (K + BK - 1) / BK;
  stageA(0, 0);
  stageB(0, 0);
  __syncthreads();

  for (int kt = 0; kt < nk; ++kt) {
    const int cur = kt & 1;

    // ---- assemble fragments (16-bit A 16x32 layout, ISA 7.12.2) ----
    FragBF aF[4], bF[2];
#pragma unroll
    for (int mt = 0; mt < 4; ++mt) {
      int row = wm * 64 + mt * 16 + ml;
#pragma unroll
      for (int j = 0; j < 8; ++j) {
        int kk = 2 * (j & 3) + ((j >> 2) * 16) + 8 * half;  // lanes16-31: +8
        aF[mt].u[j] = *(const unsigned*)&sA[cur][row * LDS_S + kk];
      }
    }
#pragma unroll
    for (int nt = 0; nt < 2; ++nt) {
      int col = wn * 32 + nt * 16 + ml;
#pragma unroll
      for (int j = 0; j < 8; ++j) {
        int kk = 2 * (j & 3) + ((j >> 2) * 16) + 8 * half;  // B: lanes16-31 +8 (same as A)
        bF[nt].u[j] = *(const unsigned*)&sB[cur][col * LDS_S + kk];
      }
    }

    // ---- stage next K-tile into the other buffer while WMMAs issue ----
    if (kt + 1 < nk) {
      stageA((kt + 1) * BK, cur ^ 1);
      stageB((kt + 1) * BK, cur ^ 1);
    }

#pragma unroll
    for (int mt = 0; mt < 4; ++mt)
#pragma unroll
      for (int nt = 0; nt < 2; ++nt)
        acc[mt][nt] = __builtin_amdgcn_wmma_f32_16x16x32_bf16(
            false, aF[mt].v, false, bF[nt].v, (short)0, acc[mt][nt], false, false);

    if (kt + 1 < nk) __syncthreads();
  }

  // ---- epilogue: 16x16 f32 C layout (VGPR r -> M=r / r+8; lane -> N) ----
  const bool rowsOK = (bm0 + BM <= M);
#pragma unroll
  for (int mt = 0; mt < 4; ++mt) {
#pragma unroll
    for (int nt = 0; nt < 2; ++nt) {
      FragF f; f.v = acc[mt][nt];
      int col  = bn0 + wn * 32 + nt * 16 + ml;
      int row0 = bm0 + wm * 64 + mt * 16 + 8 * half;
      float* p = C + (long long)row0 * N + col;
      if (rowsOK) {
        if (col < N) {                      // one divergent test per tile
#pragma unroll
          for (int r = 0; r < 8; ++r) { *p = f.f[r]; p += N; }
        }
      } else {
#pragma unroll
        for (int r = 0; r < 8; ++r) {
          if (row0 + r < M && col < N) *p = f.f[r];
          p += N;
        }
      }
    }
  }
}

// ---------------------------------------------------------------------------
// helpers / elementwise kernels
// ---------------------------------------------------------------------------
__global__ void gat_fill(float* __restrict__ p, float v, int n) {
  int i = blockIdx.x * blockDim.x + threadIdx.x;
  int s = gridDim.x * blockDim.x;
  for (; i < n; i += s) p[i] = v;
}

// a_src[n,h] = sum_c h[n,h,c]*att_src[h,c]; a_dst analogous.  One wave per (n,h)
__global__ __launch_bounds__(256)
void gat_attn_scores(const float* __restrict__ h, const float* __restrict__ att_s,
                     const float* __restrict__ att_d, float* __restrict__ a_s,
                     float* __restrict__ a_d, int N, int H, int C) {
  int w = blockIdx.x * 8 + (threadIdx.x >> 5);
  int lane = threadIdx.x & 31;
  if (w >= N * H) return;
  int n = w / H, hh = w - n * H;
  const float* row = h + (long long)n * H * C + (long long)hh * C;
  const float* as  = att_s + hh * C;
  const float* ad  = att_d + hh * C;
  float ss = 0.f, sd = 0.f;
  for (int c = lane; c < C; c += 32) {
    float v = row[c];
    ss += v * as[c];
    sd += v * ad[c];
  }
  ss = waveSum(ss);
  sd = waveSum(sd);
  if (lane == 0) { a_s[w] = ss; a_d[w] = sd; }
}

__device__ __forceinline__ void edge_sd(const int* src, const int* dst, int E,
                                        int e, int& s, int& d) {
  if (e < E) { s = src[e]; d = dst[e]; }
  else       { s = e - E; d = e - E; }      // implicit self loop
}

// pass 1: segment max of leaky_relu(a_src[s]+a_dst[d]) over dst
__global__ void gat_edge_max(const int* __restrict__ src, const int* __restrict__ dst,
                             const float* __restrict__ a_s, const float* __restrict__ a_d,
                             float* __restrict__ emax, int E, int N, int H) {
  int total = (E + N) * H;
  int i = blockIdx.x * blockDim.x + threadIdx.x;
  int stride = gridDim.x * blockDim.x;
  for (; i < total; i += stride) {
    int e = i / H, hh = i - e * H, s, d;
    edge_sd(src, dst, E, e, s, d);
    float v = a_s[s * H + hh] + a_d[d * H + hh];
    v = v > 0.f ? v : 0.2f * v;
    atomicMaxF(&emax[d * H + hh], v);
  }
}

// pass 2: e_exp per edge + segment sum
__global__ void gat_edge_exp(const int* __restrict__ src, const int* __restrict__ dst,
                             const float* __restrict__ a_s, const float* __restrict__ a_d,
                             const float* __restrict__ emax, float* __restrict__ ealpha,
                             float* __restrict__ denom, int E, int N, int H) {
  int total = (E + N) * H;
  int i = blockIdx.x * blockDim.x + threadIdx.x;
  int stride = gridDim.x * blockDim.x;
  for (; i < total; i += stride) {
    int e = i / H, hh = i - e * H, s, d;
    edge_sd(src, dst, E, e, s, d);
    float v = a_s[s * H + hh] + a_d[d * H + hh];
    v = v > 0.f ? v : 0.2f * v;
    float w = __expf(v - emax[d * H + hh]);
    ealpha[i] = w;
    atomicAdd(&denom[d * H + hh], w);
  }
}

// pass 3: agg[d] += alpha * h[s]   (one block per edge, coalesced, fp32 atomics)
__global__ __launch_bounds__(256)
void gat_edge_aggregate(const int* __restrict__ src, const int* __restrict__ dst,
                        const float* __restrict__ ealpha, const float* __restrict__ denom,
                        const float* __restrict__ h, float* __restrict__ agg,
                        int E, int N, int H, int C) {
  int e = blockIdx.x, s, d;
  edge_sd(src, dst, E, e, s, d);
  const int HC = H * C;
  const float* hrow = h + (long long)s * HC;
  float* orow       = agg + (long long)d * HC;
  if ((C & 3) == 0) {
    for (int idx = threadIdx.x * 4; idx < HC; idx += blockDim.x * 4) {
      int hh = idx / C;  // C % 4 == 0 -> head uniform across the float4
      float alpha = ealpha[(long long)e * H + hh] / (denom[d * H + hh] + 1e-16f);
      float4 v = *(const float4*)&hrow[idx];
      atomicAdd(&orow[idx + 0], alpha * v.x);
      atomicAdd(&orow[idx + 1], alpha * v.y);
      atomicAdd(&orow[idx + 2], alpha * v.z);
      atomicAdd(&orow[idx + 3], alpha * v.w);
    }
  } else {
    for (int idx = threadIdx.x; idx < HC; idx += blockDim.x) {
      int hh = idx / C;
      float alpha = ealpha[(long long)e * H + hh] / (denom[d * H + hh] + 1e-16f);
      atomicAdd(&orow[idx], alpha * hrow[idx]);
    }
  }
}

// out[i] = relu(agg[i] + b[i % F])   (F power-of-two fast path)
__global__ void gat_bias_relu(const float* __restrict__ agg, const float* __restrict__ b,
                              float* __restrict__ out, int n, int F) {
  int i = blockIdx.x * blockDim.x + threadIdx.x;
  int s = gridDim.x * blockDim.x;
  const bool pow2 = (F & (F - 1)) == 0;
  for (; i < n; i += s) {
    int c = pow2 ? (i & (F - 1)) : (i % F);
    out[i] = fmaxf(agg[i] + b[c], 0.f);
  }
}

// final: log_softmax(agg + b) per node, one wave per node (C = 42)
__global__ __launch_bounds__(256)
void gat_log_softmax(const float* __restrict__ agg, const float* __restrict__ b,
                     float* __restrict__ out, int N, int C) {
  int n = blockIdx.x * 8 + (threadIdx.x >> 5);
  int lane = threadIdx.x & 31;
  if (n >= N) return;
  const float* row = agg + (long long)n * C;
  float m = -3.4028235e38f;
  for (int c = lane; c < C; c += 32) m = fmaxf(m, row[c] + b[c]);
  m = waveMax(m);
  float s = 0.f;
  for (int c = lane; c < C; c += 32) s += __expf(row[c] + b[c] - m);
  s = waveSum(s);
  float ls = __logf(s);
  for (int c = lane; c < C; c += 32)
    out[(long long)n * C + c] = row[c] + b[c] - m - ls;
}

// ---------------------------------------------------------------------------
extern "C" void kernel_launch(void* const* d_in, const int* in_sizes, int n_in,
                              void* d_out, int out_size, void* d_ws, size_t ws_size,
                              hipStream_t stream) {
  const float* x   = (const float*)d_in[0];
  const int*   ei  = (const int*)d_in[1];
  const float* W1  = (const float*)d_in[2];
  const float* as1 = (const float*)d_in[3];
  const float* ad1 = (const float*)d_in[4];
  const float* b1  = (const float*)d_in[5];
  const float* W2  = (const float*)d_in[6];
  const float* as2 = (const float*)d_in[7];
  const float* ad2 = (const float*)d_in[8];
  const float* b2  = (const float*)d_in[9];

  const int Fin = 602, H1 = 8, C1 = 256, F1 = H1 * C1, C2 = 42;
  const int N  = in_sizes[0] / Fin;
  const int E  = in_sizes[1] / 2;
  const int EE = E + N;
  const int* srcA = ei;
  const int* dstA = ei + E;

  // workspace carve-up (256B aligned)
  char* base = (char*)d_ws;
  size_t off = 0;
  auto alloc = [&](size_t bytes) -> void* {
    void* p = base + off;
    off = (off + bytes + 255) & ~(size_t)255;
    return p;
  };
  float* h1    = (float*)alloc((size_t)N * F1 * 4);
  float* agg1  = (float*)alloc((size_t)N * F1 * 4);
  float* a_s1  = (float*)alloc((size_t)N * H1 * 4);
  float* a_d1  = (float*)alloc((size_t)N * H1 * 4);
  float* emax1 = (float*)alloc((size_t)N * H1 * 4);
  float* den1  = (float*)alloc((size_t)N * H1 * 4);
  float* eal1  = (float*)alloc((size_t)EE * H1 * 4);
  float* h2    = (float*)alloc((size_t)N * C2 * 4);
  float* agg2  = (float*)alloc((size_t)N * C2 * 4);
  float* a_s2  = (float*)alloc((size_t)N * 4);
  float* a_d2  = (float*)alloc((size_t)N * 4);
  float* emax2 = (float*)alloc((size_t)N * 4);
  float* den2  = (float*)alloc((size_t)N * 4);
  float* eal2  = (float*)alloc((size_t)EE * 4);
  (void)ws_size; (void)n_in; (void)out_size;

  dim3 blk(256);

  // ---------------- layer 1 ----------------
  gat_gemm_bf16<<<dim3((F1 + BN - 1) / BN, (N + BM - 1) / BM), blk, 0, stream>>>(
      x, W1, h1, N, F1, Fin);
  gat_attn_scores<<<(N * H1 + 7) / 8, blk, 0, stream>>>(h1, as1, ad1, a_s1, a_d1, N, H1, C1);
  gat_fill<<<1024, blk, 0, stream>>>(emax1, -3.4028235e38f, N * H1);
  gat_fill<<<1024, blk, 0, stream>>>(den1, 0.f, N * H1);
  gat_fill<<<4096, blk, 0, stream>>>(agg1, 0.f, N * F1);
  gat_edge_max<<<2048, blk, 0, stream>>>(srcA, dstA, a_s1, a_d1, emax1, E, N, H1);
  gat_edge_exp<<<2048, blk, 0, stream>>>(srcA, dstA, a_s1, a_d1, emax1, eal1, den1, E, N, H1);
  gat_edge_aggregate<<<EE, blk, 0, stream>>>(srcA, dstA, eal1, den1, h1, agg1, E, N, H1, C1);
  // relu(agg + b1) -> reuse h1 buffer as layer-2 input
  gat_bias_relu<<<4096, blk, 0, stream>>>(agg1, b1, h1, N * F1, F1);

  // ---------------- layer 2 ----------------
  gat_gemm_bf16<<<dim3((C2 + BN - 1) / BN, (N + BM - 1) / BM), blk, 0, stream>>>(
      h1, W2, h2, N, C2, F1);
  gat_attn_scores<<<(N + 7) / 8, blk, 0, stream>>>(h2, as2, ad2, a_s2, a_d2, N, 1, C2);
  gat_fill<<<512, blk, 0, stream>>>(emax2, -3.4028235e38f, N);
  gat_fill<<<512, blk, 0, stream>>>(den2, 0.f, N);
  gat_fill<<<512, blk, 0, stream>>>(agg2, 0.f, N * C2);
  gat_edge_max<<<2048, blk, 0, stream>>>(srcA, dstA, a_s2, a_d2, emax2, E, N, 1);
  gat_edge_exp<<<2048, blk, 0, stream>>>(srcA, dstA, a_s2, a_d2, emax2, eal2, den2, E, N, 1);
  gat_edge_aggregate<<<EE, 64, 0, stream>>>(srcA, dstA, eal2, den2, h2, agg2, E, N, 1, C2);

  gat_log_softmax<<<(N + 7) / 8, blk, 0, stream>>>(agg2, b2, (float*)d_out, N, C2);
}
